// DotProductAttention_49701361549674
// MI455X (gfx1250) — compile-verified, matrix operand order checked by reading, not run on hardware
//
#include <hip/hip_runtime.h>
#include <hip/hip_bf16.h>
#include <stdint.h>

typedef __bf16 bf16;
typedef __attribute__((ext_vector_type(16))) __bf16 v16bf;
typedef __attribute__((ext_vector_type(4)))  __bf16 v4bf;
typedef __attribute__((ext_vector_type(8)))  float  v8f;

#define BHN   128
#define LSEQ  1024
#define DHEAD 64
#define BQ    128    // query rows per workgroup (8 waves x 16 rows)
#define BKV   64     // key rows per iteration
#define NWAVE 8
#define KSTR  68     // padded bf16 stride for K tile rows in LDS
#define VSTR  68     // padded bf16 stride for transposed V tile
#define PSTR  68     // padded bf16 stride for P staging rows

// scores computed in log2 domain: S2 = (Q.K) * (1/sqrt(64)) * log2(e)
#define QSCALE 0.18033688011112042f           // 0.125 * log2(e)
#define MASK2  (-1442695.0409f)               // -1e6 * log2(e)

__device__ __forceinline__ float fexp2(float x) { return __builtin_amdgcn_exp2f(x); }

__launch_bounds__(256, 2)
__global__ void fa_fwd_kernel(const float* __restrict__ Q,
                              const float* __restrict__ K,
                              const float* __restrict__ V,
                              const int*   __restrict__ VLEN,
                              float* __restrict__ O) {
    __shared__ float sKf[BKV * DHEAD];      // 16 KB, raw fp32 DMA landing pad (linear)
    __shared__ bf16  sKb[BKV * KSTR];       // ~8.5 KB, bf16 K working tile (padded rows)
    __shared__ bf16  sVT[DHEAD * VSTR];     // ~8.5 KB, [d][key] bf16
    __shared__ bf16  sP[NWAVE][16 * PSTR];  // ~17 KB, per-wave P staging (C->A relayout)

    const int tid  = threadIdx.x;
    const int w    = tid >> 5;
    const int lane = tid & 31;
    const int n    = lane & 15;     // tile column for B/C/D layouts
    const int h    = lane >> 4;     // half-wave id
    const int kb0  = h ? 8 : 0;     // per-lane K-base inside A/B fragments

    const int bh   = blockIdx.x >> 3;
    const int q0   = (blockIdx.x & 7) * BQ;
    const int vlen = VLEN[bh];
    const int nblk = (vlen == 0) ? (LSEQ / BKV) : ((vlen + BKV - 1) / BKV);

    const float* Kg = K + (size_t)bh * LSEQ * DHEAD;
    const float* Vg = V + (size_t)bh * LSEQ * DHEAD;

    const uint32_t kfbase = (uint32_t)(uintptr_t)(&sKf[0]);

    // 4 per-thread async B128 DMAs for one raw K tile (linear layout)
    auto issue_k = [&](int kb) {
#pragma unroll
        for (int i = 0; i < 4; ++i) {
            const int chunk    = i * 256 + tid;          // float4 chunk id (0..1023)
            const uint32_t lds = kfbase + (uint32_t)chunk * 16u;
            const float* ga    = Kg + (size_t)kb * DHEAD + chunk * 4;
            asm volatile("global_load_async_to_lds_b128 %0, %1, off"
                         :: "v"(lds), "v"(ga) : "memory");
        }
    };
    auto load_v = [&](int kb, float4* vr) {
#pragma unroll
        for (int i = 0; i < 4; ++i) {
            const int e = (i * 256 + tid) * 4;
            vr[i] = *(const float4*)(Vg + (size_t)kb * DHEAD + e);
        }
    };

    // ---- Q A-fragments (16 rows per wave), scale*log2e folded in ----
    const float* qrow = Q + ((size_t)bh * LSEQ + q0 + w * 16 + n) * DHEAD;
    v16bf aQ[2];
#pragma unroll
    for (int c = 0; c < 2; ++c) {
#pragma unroll
        for (int j = 0; j < 8; ++j) {
            aQ[c][j]     = (bf16)(qrow[c * 32 + kb0 + j]      * QSCALE);
            aQ[c][8 + j] = (bf16)(qrow[c * 32 + kb0 + 16 + j] * QSCALE);
        }
    }

    v16bf bOne;            // all-ones B matrix: row-sums of P via WMMA
#pragma unroll
    for (int j = 0; j < 16; ++j) bOne[j] = (bf16)1.0f;

    v8f o[4];
    float m[8], l[8];
#pragma unroll
    for (int dt = 0; dt < 4; ++dt)
#pragma unroll
        for (int i = 0; i < 8; ++i) o[dt][i] = 0.0f;
#pragma unroll
    for (int i = 0; i < 8; ++i) { m[i] = -INFINITY; l[i] = 0.0f; }

    // loop-invariant LDS bases
    bf16*       pw = &sP[w][0];
    const bf16* pr = &sP[w][(lane & 15) * PSTR];

    // ---- prologue: start block 0 staging ----
    float4 vstage[4];
    issue_k(0);
    load_v(0, vstage);

    for (int blk = 0; blk < nblk; ++blk) {
        const int kb = blk * BKV;

        // ---- finish staging: convert own DMA'd K chunks fp32->bf16, store V^T ----
        asm volatile("s_wait_asynccnt 0" ::: "memory");
#pragma unroll
        for (int i = 0; i < 4; ++i) {
            const int chunk = i * 256 + tid;
            const int row = chunk >> 4, colq = chunk & 15;
            const float4 f = *(const float4*)(&sKf[chunk * 4]);
            v4bf b4;
            b4[0] = (bf16)f.x; b4[1] = (bf16)f.y;
            b4[2] = (bf16)f.z; b4[3] = (bf16)f.w;
            *(v4bf*)(&sKb[row * KSTR + colq * 4]) = b4;
        }
#pragma unroll
        for (int i = 0; i < 4; ++i) {
            const int e = (i * 256 + tid) * 4;
            const int row = e >> 6, col = e & 63;
            sVT[(col + 0) * VSTR + row] = (bf16)vstage[i].x;
            sVT[(col + 1) * VSTR + row] = (bf16)vstage[i].y;
            sVT[(col + 2) * VSTR + row] = (bf16)vstage[i].z;
            sVT[(col + 3) * VSTR + row] = (bf16)vstage[i].w;
        }
        __syncthreads();   // sKb/sVT published; sKf consumed -> free for next DMA

        // ---- prefetch next block (DMA + V regs) while we compute ----
        if (blk + 1 < nblk) {
            issue_k(kb + BKV);
            load_v(kb + BKV, vstage);
        }

        // ---- S2 = (Q*scale*log2e) K^T : 4 N-tiles x 2 K-chunks of 32 ----
        v8f s[4];
#pragma unroll
        for (int nt = 0; nt < 4; ++nt) {
            const bf16* krow = sKb + (nt * 16 + n) * KSTR;
            v8f acc;
#pragma unroll
            for (int i = 0; i < 8; ++i) acc[i] = 0.0f;
#pragma unroll
            for (int c = 0; c < 2; ++c) {
                v16bf b;
#pragma unroll
                for (int j = 0; j < 8; ++j) {
                    b[j]     = krow[c * 32 + kb0 + j];
                    b[8 + j] = krow[c * 32 + kb0 + 16 + j];
                }
                acc = __builtin_amdgcn_wmma_f32_16x16x32_bf16(
                          false, aQ[c], false, b, (short)0, acc, false, false);
            }
            s[nt] = acc;
        }

        // ---- key-padding mask: only on blocks that straddle/exceed valid_len ----
        if (kb + BKV > vlen) {
#pragma unroll
            for (int nt = 0; nt < 4; ++nt) {
                const int keyg = kb + nt * 16 + n;
                if (keyg >= vlen) {
#pragma unroll
                    for (int i = 0; i < 8; ++i) s[nt][i] = MASK2;
                }
            }
        }

        // ---- online softmax (log2 domain): max via shfl tree, sum via ones-WMMA ----
        float mnew[8], alpha[8];
#pragma unroll
        for (int i = 0; i < 8; ++i) {
            float r = fmaxf(fmaxf(s[0][i], s[1][i]), fmaxf(s[2][i], s[3][i]));
#pragma unroll
            for (int off = 1; off < 16; off <<= 1)
                r = fmaxf(r, __shfl_xor(r, off, 32));
            mnew[i]  = fmaxf(m[i], r);
            alpha[i] = fexp2(m[i] - mnew[i]);
            m[i]     = mnew[i];
        }
#pragma unroll
        for (int nt = 0; nt < 4; ++nt) {
#pragma unroll
            for (int i = 0; i < 8; ++i) {
                const float p = fexp2(s[nt][i] - mnew[i]);
                pw[(i + 8 * h) * PSTR + nt * 16 + n] = (bf16)p;
            }
        }
#pragma unroll
        for (int dt = 0; dt < 4; ++dt)
#pragma unroll
            for (int i = 0; i < 8; ++i) o[dt][i] *= alpha[i];

        asm volatile("s_wait_dscnt 0" ::: "memory");  // P writes visible to own-wave reads

        // ---- O += P V  and  l += rowsum(P) (ones-matrix WMMA) ----
        v8f rs;
#pragma unroll
        for (int i = 0; i < 8; ++i) rs[i] = 0.0f;
#pragma unroll
        for (int c = 0; c < 2; ++c) {
            v16bf aP;
#pragma unroll
            for (int j = 0; j < 8; ++j) {
                aP[j]     = pr[c * 32 + kb0 + j];
                aP[8 + j] = pr[c * 32 + kb0 + 16 + j];
            }
            rs = __builtin_amdgcn_wmma_f32_16x16x32_bf16(
                     false, aP, false, bOne, (short)0, rs, false, false);
#pragma unroll
            for (int dt = 0; dt < 4; ++dt) {
                const bf16* vrow = &sVT[(dt * 16 + n) * VSTR + c * 32];
                v16bf bV;
#pragma unroll
                for (int j = 0; j < 8; ++j) {
                    bV[j]     = vrow[kb0 + j];
                    bV[8 + j] = vrow[kb0 + 16 + j];
                }
                o[dt] = __builtin_amdgcn_wmma_f32_16x16x32_bf16(
                            false, aP, false, bV, (short)0, o[dt], false, false);
            }
        }
#pragma unroll
        for (int i = 0; i < 8; ++i) l[i] = l[i] * alpha[i] + rs[i];

        __syncthreads();  // sKb/sVT/sP reads done before next iteration's staging
    }

    // ---- epilogue: normalize and store fp32 ----
    float* obase = O + ((size_t)bh * LSEQ + q0 + w * 16) * DHEAD;
#pragma unroll
    for (int i = 0; i < 8; ++i) {
        const float inv = 1.0f / l[i];
        float* orow = obase + (i + 8 * h) * DHEAD;
#pragma unroll
        for (int dt = 0; dt < 4; ++dt)
            orow[dt * 16 + n] = o[dt][i] * inv;
    }
}

extern "C" void kernel_launch(void* const* d_in, const int* in_sizes, int n_in,
                              void* d_out, int out_size, void* d_ws, size_t ws_size,
                              hipStream_t stream) {
    (void)in_sizes; (void)n_in; (void)out_size; (void)d_ws; (void)ws_size;
    const float* Q  = (const float*)d_in[0];
    const float* K  = (const float*)d_in[1];
    const float* V  = (const float*)d_in[2];
    const int*   VL = (const int*)d_in[3];
    float* O = (float*)d_out;

    dim3 grid(BHN * (LSEQ / BQ));   // 128 * 8 = 1024 workgroups
    dim3 block(256);                // 8 wave32 waves
    fa_fwd_kernel<<<grid, block, 0, stream>>>(Q, K, V, VL, O);
}
